// UNet_59382217834756
// MI455X (gfx1250) — matrix operand order repaired and checked by reading, not compiled, not run on hardware
//
#include <hip/hip_runtime.h>
#include <hip/hip_bf16.h>

// ---------------------------------------------------------------------------
// Types for WMMA
// ---------------------------------------------------------------------------
typedef __attribute__((ext_vector_type(16))) _Float16 v16h;
typedef __attribute__((ext_vector_type(8)))  float    v8f;

#define TM 64
#define TN 64
#define TK 32
#define SA_PAD 8   // halves; row stride 80B, multiple of 16B
#define SB_PAD 8   // halves; row stride 144B, multiple of 16B
#define SA_PLANE (TM * (TK + SA_PAD))   // halves per buffer
#define SB_PLANE (TK * (TN + SB_PAD))   // halves per buffer

// ---------------------------------------------------------------------------
// GEMM: C[M,N] = act( A[M,K] @ W[K,N] + bias ),  A/W f16, C f32
// (optional extra f16 copy of C for chained GEMMs).
// Software-pipelined: double-buffered LDS; tile k+1 staged with
// GLOBAL_LOAD_ASYNC_TO_LDS_B128 (ASYNCcnt) while WMMAs run on tile k.
// One barrier per K-step. Guarded scalar staging only for ragged edges.
// 128 threads = 4 waves; each wave owns a 32x32 block (2x2 of 16x16 WMMA).
// ---------------------------------------------------------------------------
__global__ __launch_bounds__(128)
void gemm_wmma_kernel(const _Float16* __restrict__ A, int lda,
                      const _Float16* __restrict__ Wt, int ldw,
                      const float* __restrict__ bias,
                      float* __restrict__ C, int ldc,
                      _Float16* __restrict__ C16, int ldc16,
                      int M, int N, int K, int act)
{
    __shared__ __align__(16) _Float16 sA[2][TM][TK + SA_PAD];  // [buf][m][k]
    __shared__ __align__(16) _Float16 sB[2][TK][TN + SB_PAD];  // [buf][k][n]

    const int tid  = threadIdx.x;
    const int wave = tid >> 5;
    const int lane = tid & 31;
    const int bm   = blockIdx.y * TM;
    const int bn   = blockIdx.x * TN;
    const int wm   = (wave >> 1) * 32;
    const int wn   = (wave & 1) * 32;
    const int half = lane >> 4;
    const int r    = lane & 15;

    const bool fullM = (bm + TM <= M);
    const bool fullN = (bn + TN <= N);
    const bool alA   = ((lda & 7) == 0);
    const bool alB   = ((ldw & 7) == 0);

    const unsigned saBase = (unsigned)(unsigned long long)&sA[0][0][0];
    const unsigned sbBase = (unsigned)(unsigned long long)&sB[0][0][0];

    auto stageA = [&](int buf, int k0) {
        if (fullM && (k0 + TK <= K) && alA) {
#pragma unroll
            for (int i = 0; i < 2; ++i) {
                int c = tid + i * 128;          // 256 chunks of 16B
                int row = c >> 2, kc = c & 3;
                const _Float16* g = A + (long)(bm + row) * lda + k0 + kc * 8;
                unsigned l = saBase +
                    (unsigned)(buf * SA_PLANE + row * (TK + SA_PAD) + kc * 8) * 2u;
                asm volatile("global_load_async_to_lds_b128 %0, %1, off"
                             :: "v"(l), "v"(g) : "memory");
            }
        } else {
            for (int i = tid; i < TM * TK; i += 128) {
                int m = i >> 5, k = i & (TK - 1);
                int gm = bm + m, gk = k0 + k;
                sA[buf][m][k] = (gm < M && gk < K) ? A[(long)gm * lda + gk]
                                                   : (_Float16)0.0f;
            }
        }
    };
    auto stageB = [&](int buf, int k0) {
        if (fullN && (k0 + TK <= K) && alB) {
#pragma unroll
            for (int i = 0; i < 2; ++i) {
                int c = tid + i * 128;
                int row = c >> 3, nc = c & 7;
                const _Float16* g = Wt + (long)(k0 + row) * ldw + bn + nc * 8;
                unsigned l = sbBase +
                    (unsigned)(buf * SB_PLANE + row * (TN + SB_PAD) + nc * 8) * 2u;
                asm volatile("global_load_async_to_lds_b128 %0, %1, off"
                             :: "v"(l), "v"(g) : "memory");
            }
        } else {
            for (int i = tid; i < TK * TN; i += 128) {
                int k = i >> 6, n = i & (TN - 1);
                int gk = k0 + k, gn = bn + n;
                sB[buf][k][n] = (gk < K && gn < N) ? Wt[(long)gk * ldw + gn]
                                                   : (_Float16)0.0f;
            }
        }
    };

    v8f acc[2][2] = {};

    // prologue: stage tile 0
    stageA(0, 0);
    stageB(0, 0);

    int buf = 0;
    for (int k0 = 0; k0 < K; k0 += TK) {
        // my in-flight async stores (tile k) done; barrier makes every wave's
        // stores visible and proves all waves finished reading buf^1.
        asm volatile("s_wait_asynccnt 0x0" ::: "memory");
        __syncthreads();

        // kick off tile k+1 into the other buffer (overlaps with WMMAs below)
        if (k0 + TK < K) {
            stageA(buf ^ 1, k0 + TK);
            stageB(buf ^ 1, k0 + TK);
        }

        // ---- fragments: k = 16*half + s (contiguous) for both A and B ----
        v16h af[2], bf[2];
#pragma unroll
        for (int mi = 0; mi < 2; ++mi) {
            const _Float16* rowp = &sA[buf][wm + mi * 16 + r][0];
            const int ka = half * 16;
#pragma unroll
            for (int t = 0; t < 8; ++t) {
                af[mi][t]     = rowp[ka + t];
                af[mi][8 + t] = rowp[ka + 8 + t];
            }
        }
#pragma unroll
        for (int ni = 0; ni < 2; ++ni) {
            const int col = wn + ni * 16 + r;
            const int kb = half * 16;
#pragma unroll
            for (int t = 0; t < 16; ++t)
                bf[ni][t] = sB[buf][kb + t][col];
        }
#pragma unroll
        for (int mi = 0; mi < 2; ++mi)
#pragma unroll
            for (int ni = 0; ni < 2; ++ni)
                acc[mi][ni] = __builtin_amdgcn_wmma_f32_16x16x32_f16(
                    false, af[mi], false, bf[ni], (short)0, acc[mi][ni],
                    false, false);
        buf ^= 1;
    }

    // ---- store: D layout = VGPR t -> M = t (+8 upper half-wave), N = r ----
#pragma unroll
    for (int mi = 0; mi < 2; ++mi)
#pragma unroll
        for (int ni = 0; ni < 2; ++ni)
#pragma unroll
            for (int t = 0; t < 8; ++t) {
                int row = bm + wm + mi * 16 + half * 8 + t;
                int col = bn + wn + ni * 16 + r;
                if (row < M && col < N) {
                    float v = acc[mi][ni][t];
                    if (bias) v += bias[col];
                    if (act == 2) v = (v > 20.0f) ? v : log1pf(__expf(v));
                    C[(long)row * ldc + col] = v;
                    if (C16) C16[(long)row * ldc16 + col] = (_Float16)v;
                }
            }
}

// ---------------------------------------------------------------------------
__global__ void cvt_pad_kernel(const float* __restrict__ s,
                               _Float16* __restrict__ d,
                               int K, int N, int Np)
{
    long idx = (long)blockIdx.x * blockDim.x + threadIdx.x;
    long tot = (long)K * N;
    if (idx >= tot) return;
    int n = (int)(idx % N);
    long k = idx / N;
    d[k * Np + n] = (_Float16)s[idx];
}

// ---------------------------------------------------------------------------
__global__ void conv_silu_kernel(const float* __restrict__ xz,
                                 const float* __restrict__ cw,
                                 const float* __restrict__ cb,
                                 float* __restrict__ xc,
                                 _Float16* __restrict__ xc16,
                                 int Bb, int L, int E, int E8)
{
    long idx = (long)blockIdx.x * blockDim.x + threadIdx.x;
    long tot = (long)Bb * L * E;
    if (idx >= tot) return;
    int e = (int)(idx % E);
    long bl = idx / E;
    int l = (int)(bl % L);
    int b = (int)(bl / L);
    float s = cb[e];
#pragma unroll
    for (int k = 0; k < 4; ++k) {
        int ls = l + k - 3;
        if (ls >= 0)
            s += xz[((long)b * L + ls) * 2 * E + e] * cw[e * 4 + k];
    }
    float v = s / (1.0f + __expf(-s));   // silu
    xc[idx] = v;
    xc16[bl * E8 + e] = (_Float16)v;
}

// ---------------------------------------------------------------------------
__global__ void scan_kernel(const float* __restrict__ dt,
                            const float* __restrict__ xc,
                            const float* __restrict__ xz,
                            const float* __restrict__ dbl,
                            const float* __restrict__ A_log,
                            const float* __restrict__ Dp,
                            _Float16* __restrict__ y,
                            int Bb, int L, int E, int R, int E8)
{
    int idx = blockIdx.x * blockDim.x + threadIdx.x;
    if (idx >= Bb * E) return;
    int b = idx / E, e = idx % E;
    const int S = R + 32;
    float Aa[16], h[16];
#pragma unroll
    for (int n = 0; n < 16; ++n) { Aa[n] = -__expf(A_log[e * 16 + n]); h[n] = 0.0f; }
    const float Dv = Dp[e];
    for (int l = 0; l < L; ++l) {
        long base = (long)b * L + l;
        float dtv = dt[base * E + e];
        float xv  = xc[base * E + e];
        float zv  = xz[base * 2 * E + E + e];
        const float* Bm = dbl + base * S + R;
        const float* Cm = Bm + 16;
        float dx = dtv * xv;
        float yv = 0.0f;
#pragma unroll
        for (int n = 0; n < 16; ++n) {
            float dA = __expf(dtv * Aa[n]);
            h[n] = dA * h[n] + dx * Bm[n];
            yv += h[n] * Cm[n];
        }
        yv = (yv + xv * Dv) * (zv / (1.0f + __expf(-zv)));
        y[base * E8 + e] = (_Float16)yv;
    }
}

// ---------------------------------------------------------------------------
template <typename T>
__global__ __launch_bounds__(128)
void ln_relu_kernel(const float* __restrict__ x,
                    const float* __restrict__ g,
                    const float* __restrict__ bt,
                    T* __restrict__ o, int C, int ldo)
{
    long row = blockIdx.x;
    const float* xr = x + row * C;
    float s = 0.0f, ss = 0.0f;
    for (int c = threadIdx.x; c < C; c += 128) { float v = xr[c]; s += v; ss += v * v; }
    __shared__ float rs[4], rss[4];
#pragma unroll
    for (int off = 16; off > 0; off >>= 1) {
        s  += __shfl_xor(s,  off, 32);
        ss += __shfl_xor(ss, off, 32);
    }
    int lane = threadIdx.x & 31, w = threadIdx.x >> 5;
    if (lane == 0) { rs[w] = s; rss[w] = ss; }
    __syncthreads();
    s  = rs[0] + rs[1] + rs[2] + rs[3];
    ss = rss[0] + rss[1] + rss[2] + rss[3];
    float m   = s / C;
    float var = ss / C - m * m;
    float inv = rsqrtf(var + 1e-5f);
    T* orow = o + row * ldo;
    for (int c = threadIdx.x; c < C; c += 128) {
        float v = (xr[c] - m) * inv * g[c] + bt[c];
        orow[c] = (T)(v > 0.0f ? v : 0.0f);
    }
}

// ---------------------------------------------------------------------------
__global__ void chw2seq_f16_kernel(const float* __restrict__ x, _Float16* __restrict__ s,
                                   int Bb, int C, int H, int W, int Cp)
{
    long idx = (long)blockIdx.x * blockDim.x + threadIdx.x;
    long tot = (long)Bb * H * W * C;
    if (idx >= tot) return;
    int c = (int)(idx % C); long rr = idx / C;   // rr = ((b*H+h)*W+w)
    int w = (int)(rr % W);  long r2 = rr / W;
    int h = (int)(r2 % H);  int b = (int)(r2 / H);
    s[rr * Cp + c] = (_Float16)x[(((long)b * C + c) * H + h) * W + w];
}

__global__ void seq2chw_swap_kernel(const float* __restrict__ s, float* __restrict__ o,
                                    int Bb, int C, int H, int W)
{
    long idx = (long)blockIdx.x * blockDim.x + threadIdx.x;
    long tot = (long)Bb * C * W * H;
    if (idx >= tot) return;
    int h = (int)(idx % H); long r = idx / H;
    int w = (int)(r % W);   r /= W;
    int c = (int)(r % C);   int b = (int)(r / C);
    o[idx] = s[(((long)b * H + h) * W + w) * C + c];
}

__global__ void maxpool2_kernel(const float* __restrict__ x, float* __restrict__ o,
                                int Bb, int C, int H, int W)
{
    int Ho = H >> 1, Wo = W >> 1;
    long idx = (long)blockIdx.x * blockDim.x + threadIdx.x;
    long tot = (long)Bb * C * Ho * Wo;
    if (idx >= tot) return;
    int wo = (int)(idx % Wo); long r = idx / Wo;
    int ho = (int)(r % Ho);   r /= Ho;
    int c  = (int)(r % C);    int b = (int)(r / C);
    const float* base = x + (((long)b * C + c) * H + 2 * ho) * W + 2 * wo;
    o[idx] = fmaxf(fmaxf(base[0], base[1]), fmaxf(base[W], base[W + 1]));
}

__global__ void upsample2_kernel(const float* __restrict__ x, float* __restrict__ o,
                                 int Bb, int C, int H, int W)
{
    int Ho = 2 * H, Wo = 2 * W;
    long idx = (long)blockIdx.x * blockDim.x + threadIdx.x;
    long tot = (long)Bb * C * Ho * Wo;
    if (idx >= tot) return;
    int wo = (int)(idx % Wo); long r = idx / Wo;
    int ho = (int)(r % Ho);   r /= Ho;
    int c  = (int)(r % C);    int b = (int)(r / C);
    float fy = (H > 1) ? ho * (float)(H - 1) / (float)(Ho - 1) : 0.0f;
    float fx = (W > 1) ? wo * (float)(W - 1) / (float)(Wo - 1) : 0.0f;
    int y0 = (int)floorf(fy), x0 = (int)floorf(fx);
    int y1 = y0 + 1 < H ? y0 + 1 : H - 1;
    int x1 = x0 + 1 < W ? x0 + 1 : W - 1;
    float wy = fy - y0, wx = fx - x0;
    const float* bp = x + ((long)b * C + c) * H * W;
    float v00 = bp[y0 * W + x0], v01 = bp[y0 * W + x1];
    float v10 = bp[y1 * W + x0], v11 = bp[y1 * W + x1];
    o[idx] = v00 * (1 - wy) * (1 - wx) + v01 * (1 - wy) * wx
           + v10 * wy * (1 - wx)       + v11 * wy * wx;
}

__global__ void concat_kernel(const float* __restrict__ a, const float* __restrict__ b2,
                              float* __restrict__ o, int Bb, int C1, int C2, int H, int W)
{
    long idx = (long)blockIdx.x * blockDim.x + threadIdx.x;
    int Ct = C1 + C2;
    long tot = (long)Bb * Ct * H * W;
    if (idx >= tot) return;
    int hw = (int)(idx % (H * W)); long r = idx / (H * W);
    int c = (int)(r % Ct); int b = (int)(r / Ct);
    o[idx] = (c < C1) ? a[((long)b * C1 + c) * H * W + hw]
                      : b2[((long)b * C2 + (c - C1)) * H * W + hw];
}

// ---------------------------------------------------------------------------
// Host-side orchestration
// ---------------------------------------------------------------------------
namespace {

struct SSM { const float *A_log, *D, *W_dt, *W_in, *W_out, *W_x, *b_dt, *conv_b, *conv_w; };
struct LN  { const float *b, *g; };
struct Dbl { LN ln1, ln2; SSM s1, s2; };

struct Arena {
    char* base; size_t off, cap;
    void* raw(size_t bytes) {
        bytes = (bytes + 255) & ~(size_t)255;
        void* p = (void*)(base + off);
        off += bytes;
        return p;
    }
    float*    alloc(size_t n)   { return (float*)raw(n * sizeof(float)); }
    _Float16* alloc_h(size_t n) { return (_Float16*)raw(n * sizeof(_Float16)); }
};

inline int cdiv(long a, int b) { return (int)((a + b - 1) / b); }
inline int al8(int x) { return (x + 7) & ~7; }

void launch_gemm(const _Float16* A, int lda, const _Float16* W, int ldw,
                 const float* bias, float* C, int ldc,
                 _Float16* C16, int ldc16,
                 int M, int N, int K, int act, hipStream_t st)
{
    dim3 grid((N + TN - 1) / TN, (M + TM - 1) / TM, 1);
    gemm_wmma_kernel<<<grid, dim3(128), 0, st>>>(A, lda, W, ldw, bias,
                                                 C, ldc, C16, ldc16, M, N, K, act);
}

void cvt_pad(const float* s, _Float16* d, int K, int N, int Np, hipStream_t st)
{
    long tot = (long)K * N;
    cvt_pad_kernel<<<cdiv(tot, 256), 256, 0, st>>>(s, d, K, N, Np);
}

void run_mamba(const _Float16* x16, int lda, int Bb, int L, int d, int out_c,
               const SSM& P, float* out, Arena& ar, hipStream_t st)
{
    const int E = 2 * d, R = (d + 15) / 16, S = R + 32;
    const int E8 = al8(E), S8 = al8(S);
    const int N1 = 2 * E, N1p = al8(N1), ocp = al8(out_c);
    const long M = (long)Bb * L;

    _Float16* Wi  = ar.alloc_h((size_t)d * N1p);  cvt_pad(P.W_in,  Wi,  d, N1,    N1p, st);
    _Float16* Wx  = ar.alloc_h((size_t)E * S8);   cvt_pad(P.W_x,   Wx,  E, S,     S8,  st);
    _Float16* Wdt = ar.alloc_h((size_t)R * E8);   cvt_pad(P.W_dt,  Wdt, R, E,     E8,  st);
    _Float16* Wo  = ar.alloc_h((size_t)E * ocp);  cvt_pad(P.W_out, Wo,  E, out_c, ocp, st);

    float* xz = ar.alloc(M * 2 * E);
    launch_gemm(x16, lda, Wi, N1p, nullptr, xz, 2 * E, nullptr, 0,
                (int)M, 2 * E, d, 0, st);

    float* xc = ar.alloc(M * E);
    _Float16* xc16 = ar.alloc_h(M * E8);
    { long tot = M * E;
      conv_silu_kernel<<<cdiv(tot, 256), 256, 0, st>>>(xz, P.conv_w, P.conv_b,
                                                       xc, xc16, Bb, L, E, E8); }

    float* dl = ar.alloc(M * S);
    _Float16* dl16 = ar.alloc_h(M * S8);
    launch_gemm(xc16, E8, Wx, S8, nullptr, dl, S, dl16, S8,
                (int)M, S, E, 0, st);

    float* dt = ar.alloc(M * E);
    launch_gemm(dl16, S8, Wdt, E8, P.b_dt, dt, E, nullptr, 0,
                (int)M, E, R, 2, st);

    _Float16* y16 = ar.alloc_h(M * E8);
    { int tot = Bb * E;
      scan_kernel<<<cdiv(tot, 128), 128, 0, st>>>(dt, xc, xz, dl, P.A_log, P.D,
                                                  y16, Bb, L, E, R, E8); }

    launch_gemm(y16, E8, Wo, ocp, nullptr, out, out_c, nullptr, 0,
                (int)M, out_c, E, 0, st);
}

void run_double(const float* x, int Bb, int C, int H, int W,
                const Dbl& P, int mid, int oc, float* out, Arena& ar, hipStream_t st)
{
    size_t save = ar.off;
    const int L = H * W;
    const long M = (long)Bb * L;
    const int Cp = al8(C), midp = al8(mid);

    _Float16* s16 = ar.alloc_h(M * Cp);
    { long tot = M * C;
      chw2seq_f16_kernel<<<cdiv(tot, 256), 256, 0, st>>>(x, s16, Bb, C, H, W, Cp); }

    float* t = ar.alloc(M * mid);
    { size_t mk = ar.off; run_mamba(s16, Cp, Bb, L, C, mid, P.s1, t, ar, st); ar.off = mk; }

    _Float16* t2 = ar.alloc_h(M * midp);
    ln_relu_kernel<_Float16><<<(int)M, 128, 0, st>>>(t, P.ln1.g, P.ln1.b, t2, mid, midp);

    float* u = ar.alloc(M * oc);
    { size_t mk = ar.off; run_mamba(t2, midp, Bb, L, mid, oc, P.s2, u, ar, st); ar.off = mk; }

    float* u2 = ar.alloc(M * oc);
    ln_relu_kernel<float><<<(int)M, 128, 0, st>>>(u, P.ln2.g, P.ln2.b, u2, oc, oc);

    { long tot = M * oc;
      seq2chw_swap_kernel<<<cdiv(tot, 256), 256, 0, st>>>(u2, out, Bb, oc, H, W); }
    ar.off = save;
}

void run_single(const float* x, int Bb, int C, int H, int W,
                const SSM& P, int oc, float* out, Arena& ar, hipStream_t st)
{
    size_t save = ar.off;
    const int L = H * W;
    const long M = (long)Bb * L;
    const int Cp = al8(C);

    _Float16* s16 = ar.alloc_h(M * Cp);
    { long tot = M * C;
      chw2seq_f16_kernel<<<cdiv(tot, 256), 256, 0, st>>>(x, s16, Bb, C, H, W, Cp); }

    float* u = ar.alloc(M * oc);
    { size_t mk = ar.off; run_mamba(s16, Cp, Bb, L, C, oc, P, u, ar, st); ar.off = mk; }

    { long tot = M * oc;
      seq2chw_swap_kernel<<<cdiv(tot, 256), 256, 0, st>>>(u, out, Bb, oc, H, W); }
    ar.off = save;
}

} // namespace

extern "C" void kernel_launch(void* const* d_in, const int* in_sizes, int n_in,
                              void* d_out, int out_size, void* d_ws, size_t ws_size,
                              hipStream_t stream)
{
    (void)in_sizes; (void)n_in; (void)out_size; (void)ws_size;

    const float* inp = (const float*)d_in[0];
    int pi = 1;
    auto F = [&](void) { return (const float*)d_in[pi++]; };
    auto getSSM = [&](SSM& s) {
        s.A_log = F(); s.D = F(); s.W_dt = F(); s.W_in = F(); s.W_out = F();
        s.W_x = F(); s.b_dt = F(); s.conv_b = F(); s.conv_w = F();
    };
    auto getDbl = [&](Dbl& d) {
        d.ln1.b = F(); d.ln1.g = F(); d.ln2.b = F(); d.ln2.g = F();
        getSSM(d.s1); getSSM(d.s2);
    };
    Dbl down[5], incP, up[5]; SSM outcP;
    for (int i = 0; i < 5; ++i) getDbl(down[i]);
    getDbl(incP);
    getSSM(outcP);
    for (int i = 0; i < 5; ++i) getDbl(up[i]);

    Arena ar{(char*)d_ws, 0, ws_size};
    float* x0 = ar.alloc((size_t)2 * 32 * 64 * 64);
    float* x1 = ar.alloc((size_t)2 * 64 * 32 * 32);
    float* x2 = ar.alloc((size_t)2 * 128 * 16 * 16);
    float* x3 = ar.alloc((size_t)2 * 256 * 8 * 8);
    float* x4 = ar.alloc((size_t)2 * 512 * 4 * 4);
    float* x5 = ar.alloc((size_t)2 * 1024 * 2 * 2);
    float* tb = ar.alloc((size_t)2 * 64 * 64 * 64);
    float* ub = ar.alloc((size_t)2 * 64 * 64 * 64);
    float* cb = ar.alloc((size_t)2 * 96 * 64 * 64);
    float* zb = ar.alloc((size_t)2 * 32 * 64 * 64);

    const int B = 2;
    hipStream_t st = stream;

    auto pool = [&](const float* in, float* out, int C, int H, int W) {
        long tot = (long)B * C * (H / 2) * (W / 2);
        maxpool2_kernel<<<cdiv(tot, 256), 256, 0, st>>>(in, out, B, C, H, W);
    };
    auto upcat = [&](const float* a, int Ca, int Ha, int Wa,
                     const float* skip, int Cs) {
        long tu = (long)B * Ca * 2 * Ha * 2 * Wa;
        upsample2_kernel<<<cdiv(tu, 256), 256, 0, st>>>(a, ub, B, Ca, Ha, Wa);
        long tc = (long)B * (Ca + Cs) * 2 * Ha * 2 * Wa;
        concat_kernel<<<cdiv(tc, 256), 256, 0, st>>>(ub, skip, cb, B, Ca, Cs, 2 * Ha, 2 * Wa);
    };

    run_double(inp, B, 1,    64, 64, incP,    32,   32,   x0, ar, st);
    run_double(x0,  B, 32,   64, 64, down[0], 32,   64,   tb, ar, st);  pool(tb, x1, 64,   64, 64);
    run_double(x1,  B, 64,   32, 32, down[1], 64,   128,  tb, ar, st);  pool(tb, x2, 128,  32, 32);
    run_double(x2,  B, 128,  16, 16, down[2], 128,  256,  tb, ar, st);  pool(tb, x3, 256,  16, 16);
    run_double(x3,  B, 256,  8,  8,  down[3], 256,  512,  tb, ar, st);  pool(tb, x4, 512,  8,  8);
    run_double(x4,  B, 512,  4,  4,  down[4], 512,  1024, tb, ar, st);  pool(tb, x5, 1024, 4,  4);

    upcat(x5, 1024, 2,  2,  x4, 512); run_double(cb, B, 1536, 4,  4,  up[0], 1536, 512, zb, ar, st);
    upcat(zb, 512,  4,  4,  x3, 256); run_double(cb, B, 768,  8,  8,  up[1], 768,  256, zb, ar, st);
    upcat(zb, 256,  8,  8,  x2, 128); run_double(cb, B, 384,  16, 16, up[2], 384,  128, zb, ar, st);
    upcat(zb, 128,  16, 16, x1, 64);  run_double(cb, B, 192,  32, 32, up[3], 192,  64,  zb, ar, st);
    upcat(zb, 64,   32, 32, x0, 32);  run_double(cb, B, 96,   64, 64, up[4], 96,   32,  zb, ar, st);

    run_single(zb, B, 32, 64, 64, outcP, 4, (float*)d_out, ar, st);
}
